// NeuralMemory_30477087932973
// MI455X (gfx1250) — matrix operand order repaired and checked by reading, not compiled, not run on hardware
//
#include <hip/hip_runtime.h>
#include <hip/hip_bf16.h>

// ---------------------------------------------------------------------------
// NeuralMemory forward on MI455X (gfx1250), wave32 + WMMA bf16 (f32 accum).
//
// Shapes: B=32, T=2048, C=768, H=1536, L=16, window W=T/L=128.
//   (pre-pass) x,Wq,W0,W1 -> bf16 in workspace (one-shot, memory-bound)
//   q   = x @ Wq^T + bq                       [B*T, C]   (bf16 GEMM, bf16 out)
//   h   = silu(q @ W0[b]^T + b0[b])           [B*T, H]   (bf16 GEMM, bf16 out)
//   mem = h @ W1[b]^T + b1[b]  --(pool 128)-> pooled [B*L, C] (atomic f32)
//   out = pooled @ Wo^T + bo                  [B*L, C]   (f32, d_out)
//
// GEMM waves compute 64x64 tiles (16 WMMA : 16 b128-loads per K=32 step).
// Blocks map 8 waves to 8 consecutive M-tiles sharing one N-block, so the
// B panel is fetched once into WGP$ and reused by all 8 waves.
// ---------------------------------------------------------------------------

typedef __attribute__((ext_vector_type(16))) __bf16 v16bf;
typedef __attribute__((ext_vector_type(8)))  __bf16 v8bf;
typedef __attribute__((ext_vector_type(8)))  float  v8f;

#define NM_B 32
#define NM_T 2048
#define NM_C 768
#define NM_H 1536
#define NM_L 16
#define NM_WIN 128               // T / L
#define NM_MT (NM_B * NM_T)      // 65536 rows total

__device__ __forceinline__ v8f wmma_bf16(v16bf a, v16bf b, v8f c) {
    return __builtin_amdgcn_wmma_f32_16x16x32_bf16(
        /*neg_a=*/false, a, /*neg_b=*/false, b,
        /*c_mod=*/(short)0, c, /*reuse_a=*/false, /*reuse_b=*/false);
}

__device__ __forceinline__ void pack8(v16bf& r, int base, float4 a, float4 b) {
    r[base + 0] = (__bf16)a.x; r[base + 1] = (__bf16)a.y;
    r[base + 2] = (__bf16)a.z; r[base + 3] = (__bf16)a.w;
    r[base + 4] = (__bf16)b.x; r[base + 5] = (__bf16)b.y;
    r[base + 6] = (__bf16)b.z; r[base + 7] = (__bf16)b.w;
}

// A fragment (16x32 bf16): lane holds row m0+(lane&15); two contiguous 8-half
// chunks at K offsets (lane<16?0:8) and 16+(lane<16?0:8).
__device__ __forceinline__ v16bf load_a_bf16(const __bf16* __restrict__ A, long lda,
                                             long m0, int k, int lane) {
    long m = m0 + (lane & 15);
    int off = (lane < 16) ? 0 : 8;
    const v8bf* p0 = (const v8bf*)(A + m * lda + k + off);
    const v8bf* p1 = (const v8bf*)(A + m * lda + k + off + 16);
    v8bf lo = *p0, hi = *p1;
    v16bf r;
#pragma unroll
    for (int i = 0; i < 8; ++i) { r[i] = lo[i]; r[i + 8] = hi[i]; }
    return r;
}

// B fragment (32x16 bf16): lane holds column n0+(lane&15) = weight row n;
// two contiguous 8-half chunks at K offsets (lane<16?0:8) and 16+(lane<16?0:8).
__device__ __forceinline__ v16bf load_b_bf16(const __bf16* __restrict__ Bm, long ldb,
                                             int n0, int k, int lane) {
    long n = n0 + (lane & 15);
    int koff = (lane < 16) ? 0 : 8;
    const v8bf* p0 = (const v8bf*)(Bm + n * ldb + k + koff);
    const v8bf* p1 = (const v8bf*)(Bm + n * ldb + k + koff + 16);
    v8bf lo = *p0, hi = *p1;
    v16bf r;
#pragma unroll
    for (int i = 0; i < 8; ++i) { r[i] = lo[i]; r[i + 8] = hi[i]; }
    return r;
}

// f32-source fragments (only used by the tiny final GEMM)
__device__ __forceinline__ v16bf load_a_f32(const float* __restrict__ A, long lda,
                                            long m0, int k, int lane) {
    long m = m0 + (lane & 15);
    int off = (lane < 16) ? 0 : 8;
    const float* p = A + m * lda + k + off;
    const float4* q0 = (const float4*)(p);
    const float4* q1 = (const float4*)(p + 16);
    v16bf r;
    pack8(r, 0, q0[0], q0[1]);
    pack8(r, 8, q1[0], q1[1]);
    return r;
}

__device__ __forceinline__ v16bf load_b_f32(const float* __restrict__ Bm, long ldb,
                                            int n0, int k, int lane) {
    long n = n0 + (lane & 15);
    int koff = (lane < 16) ? 0 : 8;
    const float4* q = (const float4*)(Bm + n * ldb + k + koff);
    v16bf r;
    pack8(r, 0, q[0], q[1]);
    pack8(r, 8, q[4], q[5]);
    return r;
}

// ---------------------------------------------------------------------------
// Pre-pass: elementwise f32 -> bf16 (8 elements / thread, 32B in / 16B out)
// ---------------------------------------------------------------------------
__global__ __launch_bounds__(256)
void nm_cvt_bf16(const float* __restrict__ src, __bf16* __restrict__ dst, long n8) {
    long i = (long)blockIdx.x * blockDim.x + threadIdx.x;
    if (i >= n8) return;
    const float4* p = (const float4*)(src + i * 8);
    float4 a = p[0], b = p[1];
    v8bf o;
    o[0] = (__bf16)a.x; o[1] = (__bf16)a.y; o[2] = (__bf16)a.z; o[3] = (__bf16)a.w;
    o[4] = (__bf16)b.x; o[5] = (__bf16)b.y; o[6] = (__bf16)b.z; o[7] = (__bf16)b.w;
    *(v8bf*)(dst + i * 8) = o;
}

// ---------------------------------------------------------------------------
// Kernel 0: zero the pooled accumulator
// ---------------------------------------------------------------------------
__global__ void nm_zero_pooled(float* __restrict__ pooled, int n) {
    int i = blockIdx.x * blockDim.x + threadIdx.x;
    if (i < n) pooled[i] = 0.0f;
}

// ---------------------------------------------------------------------------
// Kernel 1: q = x @ Wq^T + bq       (bf16 GEMM)   M=65536 N=768 K=768
// 64x64 per wave; block = 8 waves on 8 consecutive M-tiles, same N-block.
// ---------------------------------------------------------------------------
__global__ __launch_bounds__(256)
void nm_gemm_q(const __bf16* __restrict__ X, const __bf16* __restrict__ Wq,
               const float* __restrict__ bq, __bf16* __restrict__ Q) {
    const int lane = threadIdx.x & 31;
    const int wv   = threadIdx.x >> 5;
    const int nblocks = NM_C / 64;                 // 12
    const int nb  = blockIdx.x % nblocks;
    const int mtg = blockIdx.x / nblocks;
    const long mt = (long)mtg * 8 + wv;            // 64-row tile index
    if (mt >= NM_MT / 64) return;
    const long m0 = mt * 64;
    const int  n0 = nb * 64;

    v8f acc[4][4];
#pragma unroll
    for (int i = 0; i < 4; ++i)
#pragma unroll
        for (int j = 0; j < 4; ++j) acc[i][j] = (v8f)(0.0f);

    for (int k = 0; k < NM_C; k += 32) {
        v16bf a[4];
#pragma unroll
        for (int i = 0; i < 4; ++i) a[i] = load_a_bf16(X, NM_C, m0 + 16 * i, k, lane);
#pragma unroll
        for (int j = 0; j < 4; ++j) {
            v16bf b = load_b_bf16(Wq, NM_C, n0 + 16 * j, k, lane);
#pragma unroll
            for (int i = 0; i < 4; ++i) acc[i][j] = wmma_bf16(a[i], b, acc[i][j]);
        }
    }

    const int ncol = lane & 15;
    const int roff = (lane < 16) ? 0 : 8;
#pragma unroll
    for (int i = 0; i < 4; ++i)
#pragma unroll
        for (int j = 0; j < 4; ++j) {
            int n = n0 + 16 * j + ncol;
            float bias = bq[n];
#pragma unroll
            for (int r = 0; r < 8; ++r) {
                long m = m0 + 16 * i + r + roff;
                Q[m * NM_C + n] = (__bf16)(acc[i][j][r] + bias);
            }
        }
}

// ---------------------------------------------------------------------------
// Kernel 2: h = silu(q @ W0[b]^T + b0[b])   M=65536 N=1536 K=768
// ---------------------------------------------------------------------------
__global__ __launch_bounds__(256)
void nm_gemm_h(const __bf16* __restrict__ Q, const __bf16* __restrict__ W0,
               const float* __restrict__ b0, __bf16* __restrict__ Hb) {
    const int lane = threadIdx.x & 31;
    const int wv   = threadIdx.x >> 5;
    const int nblocks = NM_H / 64;                 // 24
    const int nb  = blockIdx.x % nblocks;
    const int mtg = blockIdx.x / nblocks;
    const long mt = (long)mtg * 8 + wv;
    if (mt >= NM_MT / 64) return;
    const long m0 = mt * 64;
    const int  n0 = nb * 64;
    const int  b  = (int)(m0 >> 11);               // m0 / T (2048 % 64 == 0)
    const __bf16* W0b = W0 + (long)b * NM_H * NM_C;
    const float* bias0 = b0 + (long)b * NM_H;

    v8f acc[4][4];
#pragma unroll
    for (int i = 0; i < 4; ++i)
#pragma unroll
        for (int j = 0; j < 4; ++j) acc[i][j] = (v8f)(0.0f);

    for (int k = 0; k < NM_C; k += 32) {
        v16bf a[4];
#pragma unroll
        for (int i = 0; i < 4; ++i) a[i] = load_a_bf16(Q, NM_C, m0 + 16 * i, k, lane);
#pragma unroll
        for (int j = 0; j < 4; ++j) {
            v16bf bfr = load_b_bf16(W0b, NM_C, n0 + 16 * j, k, lane);
#pragma unroll
            for (int i = 0; i < 4; ++i) acc[i][j] = wmma_bf16(a[i], bfr, acc[i][j]);
        }
    }

    const int ncol = lane & 15;
    const int roff = (lane < 16) ? 0 : 8;
#pragma unroll
    for (int i = 0; i < 4; ++i)
#pragma unroll
        for (int j = 0; j < 4; ++j) {
            int n = n0 + 16 * j + ncol;
            float bias = bias0[n];
#pragma unroll
            for (int r = 0; r < 8; ++r) {
                long m = m0 + 16 * i + r + roff;
                float v = acc[i][j][r] + bias;
                v = v / (1.0f + __expf(-v));       // SiLU
                Hb[m * NM_H + n] = (__bf16)v;
            }
        }
}

// ---------------------------------------------------------------------------
// Kernel 3: mem = h @ W1[b]^T + b1[b], fused adaptive-avg-pool into pooled
//           M=65536 N=768 K=1536; each 64-row tile lies in one 128-row window
// ---------------------------------------------------------------------------
__global__ __launch_bounds__(256)
void nm_gemm_mem(const __bf16* __restrict__ Hb, const __bf16* __restrict__ W1,
                 const float* __restrict__ b1, float* __restrict__ pooled) {
    const int lane = threadIdx.x & 31;
    const int wv   = threadIdx.x >> 5;
    const int nblocks = NM_C / 64;                 // 12
    const int nb  = blockIdx.x % nblocks;
    const int mtg = blockIdx.x / nblocks;
    const long mt = (long)mtg * 8 + wv;
    if (mt >= NM_MT / 64) return;
    const long m0 = mt * 64;
    const int  n0 = nb * 64;
    const int  b  = (int)(m0 >> 11);
    const __bf16* W1b = W1 + (long)b * NM_C * NM_H;
    const float* bias1 = b1 + (long)b * NM_C;

    v8f acc[4][4];
#pragma unroll
    for (int i = 0; i < 4; ++i)
#pragma unroll
        for (int j = 0; j < 4; ++j) acc[i][j] = (v8f)(0.0f);

    for (int k = 0; k < NM_H; k += 32) {
        v16bf a[4];
#pragma unroll
        for (int i = 0; i < 4; ++i) a[i] = load_a_bf16(Hb, NM_H, m0 + 16 * i, k, lane);
#pragma unroll
        for (int j = 0; j < 4; ++j) {
            v16bf bfr = load_b_bf16(W1b, NM_H, n0 + 16 * j, k, lane);
#pragma unroll
            for (int i = 0; i < 4; ++i) acc[i][j] = wmma_bf16(a[i], bfr, acc[i][j]);
        }
    }

    // Pooling epilogue: column sums over the 64 rows of this tile, then
    // cross-half reduce and atomically accumulate mean into pooled[b, l, n].
    const int ncol = lane & 15;
    const int tloc = (int)(m0 & (NM_T - 1));
    const int l = tloc >> 7;                       // / 128 (64-row tile in one window)
    float* pw = pooled + (((long)b * NM_L + l) * NM_C);
#pragma unroll
    for (int j = 0; j < 4; ++j) {
        float s = 0.0f;
#pragma unroll
        for (int i = 0; i < 4; ++i)
#pragma unroll
            for (int r = 0; r < 8; ++r) s += acc[i][j][r];
        s += __shfl_xor(s, 16);
        int n = n0 + 16 * j + ncol;
        float contrib = (s + 64.0f * bias1[n]) * (1.0f / (float)NM_WIN);
        if (lane < 16) atomicAdd(pw + n, contrib);
    }
}

// ---------------------------------------------------------------------------
// Kernel 4: out = pooled @ Wo^T + bo     M=512 N=768 K=768   (f32 out)
// ---------------------------------------------------------------------------
__global__ __launch_bounds__(256)
void nm_gemm_out(const float* __restrict__ P, const float* __restrict__ Wo,
                 const float* __restrict__ bo, float* __restrict__ Out) {
    const int lane = threadIdx.x & 31;
    const int wv   = threadIdx.x >> 5;
    const int nblocks = NM_C / 64;                 // 12
    const int nb  = blockIdx.x % nblocks;
    const int mtg = blockIdx.x / nblocks;
    const long mt = (long)mtg * 8 + wv;            // 8 M-tiles of 64 rows total
    if (mt >= (NM_B * NM_L) / 64) return;
    const long m0 = mt * 64;
    const int  n0 = nb * 64;

    v8f acc[4][4];
#pragma unroll
    for (int i = 0; i < 4; ++i)
#pragma unroll
        for (int j = 0; j < 4; ++j) acc[i][j] = (v8f)(0.0f);

    for (int k = 0; k < NM_C; k += 32) {
        v16bf a[4];
#pragma unroll
        for (int i = 0; i < 4; ++i) a[i] = load_a_f32(P, NM_C, m0 + 16 * i, k, lane);
#pragma unroll
        for (int j = 0; j < 4; ++j) {
            v16bf b = load_b_f32(Wo, NM_C, n0 + 16 * j, k, lane);
#pragma unroll
            for (int i = 0; i < 4; ++i) acc[i][j] = wmma_bf16(a[i], b, acc[i][j]);
        }
    }

    const int ncol = lane & 15;
    const int roff = (lane < 16) ? 0 : 8;
#pragma unroll
    for (int i = 0; i < 4; ++i)
#pragma unroll
        for (int j = 0; j < 4; ++j) {
            int n = n0 + 16 * j + ncol;
            float bias = bo[n];
#pragma unroll
            for (int r = 0; r < 8; ++r) {
                long m = m0 + 16 * i + r + roff;
                Out[m * NM_C + n] = acc[i][j][r] + bias;
            }
        }
}

// ---------------------------------------------------------------------------
// Host launcher
// ---------------------------------------------------------------------------
extern "C" void kernel_launch(void* const* d_in, const int* in_sizes, int n_in,
                              void* d_out, int out_size, void* d_ws, size_t ws_size,
                              hipStream_t stream) {
    const float* x  = (const float*)d_in[0];
    const float* Wq = (const float*)d_in[1];
    const float* bq = (const float*)d_in[2];
    const float* W0 = (const float*)d_in[3];
    const float* b0 = (const float*)d_in[4];
    const float* W1 = (const float*)d_in[5];
    const float* b1 = (const float*)d_in[6];
    const float* Wo = (const float*)d_in[7];
    const float* bo = (const float*)d_in[8];
    float* out = (float*)d_out;

    // workspace layout (bytes, all 16B aligned)
    const size_t XB_BYTES  = (size_t)NM_MT * NM_C * 2;          // 100,663,296
    const size_t QB_BYTES  = (size_t)NM_MT * NM_C * 2;          // 100,663,296
    const size_t HB_BYTES  = (size_t)NM_MT * NM_H * 2;          // 201,326,592
    const size_t W0B_BYTES = (size_t)NM_B * NM_H * NM_C * 2;    //  75,497,472
    const size_t W1B_BYTES = (size_t)NM_B * NM_C * NM_H * 2;    //  75,497,472
    const size_t WQB_BYTES = (size_t)NM_C * NM_C * 2;           //   1,179,648

    char* base = (char*)d_ws;
    __bf16* xb  = (__bf16*)(base);                      base += XB_BYTES;
    __bf16* qb  = (__bf16*)(base);                      base += QB_BYTES;
    __bf16* hb  = (__bf16*)(base);                      base += HB_BYTES;
    __bf16* w0b = (__bf16*)(base);                      base += W0B_BYTES;
    __bf16* w1b = (__bf16*)(base);                      base += W1B_BYTES;
    __bf16* wqb = (__bf16*)(base);                      base += WQB_BYTES;
    float*  pooled = (float*)(base);

    // ---- pre-pass conversions (f32 -> bf16), 8 elements / thread ----
    {
        long n8;
        n8 = (long)NM_MT * NM_C / 8;            // x: 6,291,456
        nm_cvt_bf16<<<(int)((n8 + 255) / 256), 256, 0, stream>>>(x, xb, n8);
        n8 = (long)NM_C * NM_C / 8;             // Wq: 73,728
        nm_cvt_bf16<<<(int)((n8 + 255) / 256), 256, 0, stream>>>(Wq, wqb, n8);
        n8 = (long)NM_B * NM_H * NM_C / 8;      // W0: 4,718,592
        nm_cvt_bf16<<<(int)((n8 + 255) / 256), 256, 0, stream>>>(W0, w0b, n8);
        n8 = (long)NM_B * NM_C * NM_H / 8;      // W1: 4,718,592
        nm_cvt_bf16<<<(int)((n8 + 255) / 256), 256, 0, stream>>>(W1, w1b, n8);
    }

    const int pooled_elems = NM_B * NM_L * NM_C;        // 393216
    nm_zero_pooled<<<(pooled_elems + 255) / 256, 256, 0, stream>>>(pooled, pooled_elems);

    // 1) q projection: (65536/64 / 8) * 12 = 1536 blocks
    nm_gemm_q<<<1536, 256, 0, stream>>>(xb, wqb, bq, qb);

    // 2) hidden: (1024/8) * 24 = 3072 blocks
    nm_gemm_h<<<3072, 256, 0, stream>>>(qb, w0b, b0, hb);

    // 3) mem + pooling: (1024/8) * 12 = 1536 blocks
    nm_gemm_mem<<<1536, 256, 0, stream>>>(hb, w1b, b1, pooled);

    // 4) output projection: (8/8) * 12 = 12 blocks
    nm_gemm_out<<<12, 256, 0, stream>>>(pooled, Wo, bo, out);
}
